// Bank_25821343383842
// MI455X (gfx1250) — matrix operand order repaired and checked by reading, not compiled, not run on hardware
//
#include <hip/hip_runtime.h>
#include <hip/hip_bf16.h>
#include <stdint.h>

typedef __attribute__((ext_vector_type(16))) _Float16     v16h;
typedef __attribute__((ext_vector_type(8)))  _Float16     v8h;
typedef __attribute__((ext_vector_type(8)))  float        v8f;
typedef __attribute__((ext_vector_type(4)))  unsigned int v4u;
typedef __attribute__((ext_vector_type(8)))  int          v8i;
typedef __attribute__((ext_vector_type(4)))  int          v4i;

#define BATCH 8
#define CH    64
#define TM    256   // output pixels per workgroup (8 waves x 32 rows)
#define KC    64    // K chunk per iteration == one filter tap (Cin = 64)

// ---- Tensor Data Mover availability (builtin arity differs per toolchain) ----
#if defined(__has_builtin)
#  if __has_builtin(__builtin_amdgcn_tensor_load_to_lds) && __has_builtin(__builtin_amdgcn_s_wait_tensorcnt)
#    define HAVE_TDM 1
#  else
#    define HAVE_TDM 0
#  endif
#else
#  define HAVE_TDM 0
#endif
#if __has_include(<hip/amd_detail/amd_gfx1250_TDM.h>)
#  define TDM_ARITY6 1
#else
#  define TDM_ARITY6 0
#endif

#if defined(__HIP_DEVICE_COMPILE__)
#  if HAVE_TDM
#    warning "CDNA5: tensor_load_to_lds builtin AVAILABLE (TDM path enabled)"
#  else
#    warning "CDNA5: tensor_load_to_lds builtin NOT available (fallback B staging)"
#  endif
#endif

#if HAVE_TDM
// 2-D f16 tile load Global->LDS via TDM, per CDNA5 D# Group0/Group1 layout.
__device__ __forceinline__ void tdm_load_2d_f16(unsigned lds_addr, const void* gptr,
                                                unsigned tensor_d0, unsigned tensor_d1,
                                                unsigned tile_d0, unsigned tile_d1,
                                                unsigned stride0 /* elements */)
{
    const unsigned long long ga = (unsigned long long)(uintptr_t)gptr;
    v4u g0;
    g0[0] = 1u;                                                // count=1 (valid user D#)
    g0[1] = lds_addr;                                          // lds_addr [63:32]
    g0[2] = (unsigned)(ga & 0xffffffffu);                      // global_addr lo
    g0[3] = (unsigned)((ga >> 32) & 0x01ffffffu) | (2u << 30); // global_addr hi | type=2
    v8i g1;
    g1[0] = (int)(1u << 16);                                   // data_size=1 (2 bytes)
    g1[1] = (int)((tensor_d0 & 0xffffu) << 16);                // tensor_dim0[15:0]
    g1[2] = (int)((tensor_d0 >> 16) | ((tensor_d1 & 0xffffu) << 16));
    g1[3] = (int)((tensor_d1 >> 16) | (tile_d0 << 16));        // tile_dim0
    g1[4] = (int)(tile_d1 & 0xffffu);                          // tile_dim1 (tile_dim2=0)
    g1[5] = (int)stride0;                                      // tensor_dim0_stride[31:0]
    g1[6] = 0;                                                 // stride0[47:32]|stride1[15:0]
    g1[7] = 0;
    v4i gz = {0, 0, 0, 0};
#if TDM_ARITY6
    v8i gz8 = {0, 0, 0, 0, 0, 0, 0, 0};
    __builtin_amdgcn_tensor_load_to_lds(g0, g1, gz, gz, gz8, 0);
#else
    __builtin_amdgcn_tensor_load_to_lds(g0, g1, gz, gz, 0);
#endif
}
#endif

__device__ __forceinline__ void wait_asynccnt0()
{
#if defined(__HIP_DEVICE_COMPILE__)
    asm volatile("s_wait_asynccnt 0x0" ::: "memory");
#endif
}

// ---------------------------------------------------------------------------
// Implicit-GEMM convolution, f16 in / f32 WMMA accumulate / f16 out, NHWC.
// Per workgroup: 256 output pixels x 64 channels. Per wave: 32 pixels.
// Per K-iteration (KC=64 == one filter tap): 16x v_wmma_f32_16x16x32_f16.
// Double-buffered LDS; A tile staged with GLOBAL_LOAD_ASYNC_TO_LDS_B128,
// B (weight) tile staged with TENSOR_LOAD_TO_LDS when available.
// ---------------------------------------------------------------------------
__global__ __launch_bounds__(256)
void conv_wmma_kernel(const _Float16* __restrict__ in,   // [B,Hi,Wi,64]
                      const _Float16* __restrict__ wT,   // [64][K], K=(ky*kw+kx)*64+ci
                      const float*    __restrict__ bias, // [64]
                      _Float16*       __restrict__ out,  // [B,Ho,Wo,64]
                      int Hi, int Wi, int Ho, int Wo,
                      int kh, int kw, int stride, int pad)
{
    __shared__ alignas(64) _Float16 Atile[2][TM][KC];  // 64 KB
    __shared__ alignas(64) _Float16 Btile[2][CH][KC];  // 16 KB

    const int K    = kh * kw * CH;
    const int b    = blockIdx.y;
    const int npix = Ho * Wo;
    const int pix0 = blockIdx.x * TM;
    const int tid  = threadIdx.x;
    const int lane = tid & 31;
    const int wave = tid >> 5;

    // A staging: thread t stages the full 64-K chunk (128 B) of pixel row t
    const int  arow    = tid;
    const int  apix    = pix0 + arow;
    const int  aho     = apix / Wo;
    const int  awo     = apix - aho * Wo;
    const bool apix_ok = (apix < npix);

#if HAVE_TDM
    const bool tdm_wave = (wave == 0);
#else
    const int bcol = tid & 63;
    const int bjb  = (tid >> 6) << 4;   // 0,16,32,48
#endif

    // ---- stage helpers (double-buffered) ----
    auto stageA = [&](int kc, int bufi) {
        const int tap = kc >> 6;            // ky*kw + kx
        const int ky  = tap / kw;
        const int kx  = tap - ky * kw;
        const int hi  = aho * stride - pad + ky;
        const int wi  = awo * stride - pad + kx;
        const bool valid =
            apix_ok && (unsigned)hi < (unsigned)Hi && (unsigned)wi < (unsigned)Wi;
        if (valid) {
#if defined(__HIP_DEVICE_COMPILE__)
            const unsigned lds0 = (unsigned)(uintptr_t)&Atile[bufi][arow][0];
            const unsigned long long ga = (unsigned long long)(uintptr_t)
                (in + ((((size_t)b * Hi + hi) * Wi + wi) << 6));
            // INST_OFFSET is added to both the LDS and the global address.
            asm volatile(
                "global_load_async_to_lds_b128 %0, %1, off offset:0\n\t"
                "global_load_async_to_lds_b128 %0, %1, off offset:16\n\t"
                "global_load_async_to_lds_b128 %0, %1, off offset:32\n\t"
                "global_load_async_to_lds_b128 %0, %1, off offset:48\n\t"
                "global_load_async_to_lds_b128 %0, %1, off offset:64\n\t"
                "global_load_async_to_lds_b128 %0, %1, off offset:80\n\t"
                "global_load_async_to_lds_b128 %0, %1, off offset:96\n\t"
                "global_load_async_to_lds_b128 %0, %1, off offset:112"
                :: "v"(lds0), "v"(ga) : "memory");
#endif
        } else {
            const float4 z   = make_float4(0.f, 0.f, 0.f, 0.f);
            float4*      dst = (float4*)&Atile[bufi][arow][0];
#pragma unroll
            for (int q = 0; q < 8; ++q) dst[q] = z;
        }
    };
    auto stageB = [&](int kc, int bufi) {
#if HAVE_TDM
        if (tdm_wave)
            tdm_load_2d_f16((unsigned)(uintptr_t)&Btile[bufi][0][0], wT + kc,
                            (unsigned)K, CH, KC, CH, (unsigned)K);
#else
        const float4* src = (const float4*)(wT + (size_t)bcol * K + kc + bjb);
        float4 w0 = src[0], w1 = src[1];
        *(float4*)&Btile[bufi][bcol][bjb]     = w0;
        *(float4*)&Btile[bufi][bcol][bjb + 8] = w1;
#endif
    };
    auto waitStages = [&]() {
        wait_asynccnt0();
#if HAVE_TDM
        if (tdm_wave) __builtin_amdgcn_s_wait_tensorcnt(0);
#endif
        __syncthreads();
    };

    v8f acc00 = {}, acc01 = {}, acc02 = {}, acc03 = {};
    v8f acc10 = {}, acc11 = {}, acc12 = {}, acc13 = {};

    // ---- prologue: fill buffer 0 ----
    stageB(0, 0);
    stageA(0, 0);
    waitStages();

    for (int kc = 0; kc < K; kc += KC) {
        const int cur = (kc >> 6) & 1;
        const int nxt = cur ^ 1;
        // ---- prefetch next chunk into the other buffer (overlaps with WMMA) ----
        if (kc + KC < K) {
            stageB(kc + KC, nxt);
            stageA(kc + KC, nxt);
        }
        // ---- 16 WMMAs: 2 M-tiles x 4 N-tiles x 2 K-subchunks ----
        {
            const int row0 = (wave << 5) + (lane & 15);
            const int half = lane >> 4;
            const int cb0  = lane & 15;
            const int koff = half << 4;   // lanes 0-15: K0-15, lanes 16-31: K16-31
#pragma unroll
            for (int kk2 = 0; kk2 < 2; ++kk2) {
                const int kb = kk2 << 5;
                // A lane layout: lanes 0-15 hold K0-15, 16-31 hold K16-31
                v8h a0lo = *(const v8h*)&Atile[cur][row0][kb + (half << 4)];
                v8h a0hi = *(const v8h*)&Atile[cur][row0][kb + 8 + (half << 4)];
                v16h a0  = __builtin_shufflevector(a0lo, a0hi,
                             0, 1, 2, 3, 4, 5, 6, 7, 8, 9, 10, 11, 12, 13, 14, 15);
                v8h a1lo = *(const v8h*)&Atile[cur][row0 + 16][kb + (half << 4)];
                v8h a1hi = *(const v8h*)&Atile[cur][row0 + 16][kb + 8 + (half << 4)];
                v16h a1  = __builtin_shufflevector(a1lo, a1hi,
                             0, 1, 2, 3, 4, 5, 6, 7, 8, 9, 10, 11, 12, 13, 14, 15);
                v16h b0 = *(const v16h*)&Btile[cur][cb0     ][kb + koff];
                v16h b1 = *(const v16h*)&Btile[cur][cb0 + 16][kb + koff];
                v16h b2 = *(const v16h*)&Btile[cur][cb0 + 32][kb + koff];
                v16h b3 = *(const v16h*)&Btile[cur][cb0 + 48][kb + koff];
                acc00 = __builtin_amdgcn_wmma_f32_16x16x32_f16(false, a0, false, b0, (short)0, acc00, false, false);
                acc01 = __builtin_amdgcn_wmma_f32_16x16x32_f16(false, a0, false, b1, (short)0, acc01, false, false);
                acc02 = __builtin_amdgcn_wmma_f32_16x16x32_f16(false, a0, false, b2, (short)0, acc02, false, false);
                acc03 = __builtin_amdgcn_wmma_f32_16x16x32_f16(false, a0, false, b3, (short)0, acc03, false, false);
                acc10 = __builtin_amdgcn_wmma_f32_16x16x32_f16(false, a1, false, b0, (short)0, acc10, false, false);
                acc11 = __builtin_amdgcn_wmma_f32_16x16x32_f16(false, a1, false, b1, (short)0, acc11, false, false);
                acc12 = __builtin_amdgcn_wmma_f32_16x16x32_f16(false, a1, false, b2, (short)0, acc12, false, false);
                acc13 = __builtin_amdgcn_wmma_f32_16x16x32_f16(false, a1, false, b3, (short)0, acc13, false, false);
            }
        }
        // one barrier per iteration: publishes nxt, releases cur for overwrite
        waitStages();
    }

    // ---- epilogue: D layout = lanes 0-15 N=lane/M=v, lanes 16-31 N=lane-16/M=8+v ----
    const int    chan  = lane & 15;
    const int    prowb = (wave << 5) + ((lane >> 4) << 3);
    const size_t obase = ((size_t)b * npix) << 6;
#define CDNA5_EPI(MT, NT, ACC)                                                    \
    {                                                                             \
        const int   c  = ((NT) << 4) + chan;                                      \
        const float bv = bias[c];                                                 \
        _Pragma("unroll")                                                         \
        for (int v = 0; v < 8; ++v) {                                             \
            const int pix = pix0 + prowb + ((MT) << 4) + v;                       \
            if (pix < npix)                                                       \
                out[obase + ((size_t)pix << 6) + c] = (_Float16)((ACC)[v] + bv);  \
        }                                                                         \
    }
    CDNA5_EPI(0, 0, acc00)
    CDNA5_EPI(0, 1, acc01)
    CDNA5_EPI(0, 2, acc02)
    CDNA5_EPI(0, 3, acc03)
    CDNA5_EPI(1, 0, acc10)
    CDNA5_EPI(1, 1, acc11)
    CDNA5_EPI(1, 2, acc12)
    CDNA5_EPI(1, 3, acc13)
#undef CDNA5_EPI
}

// ---------------------------------------------------------------------------
// GroupNorm statistics: one block per (batch, group); mean + rsqrt(var+eps)
// ---------------------------------------------------------------------------
__global__ __launch_bounds__(256)
void gn_stats_kernel(const _Float16* __restrict__ x, float* __restrict__ stats, int HW)
{
    const int bg = blockIdx.x;      // b*32 + g
    const int g  = bg & 31;
    const int b  = bg >> 5;
    const int c0 = g << 1;
    const _Float16* base = x + (((size_t)b * HW) << 6);
    float s = 0.f, s2 = 0.f;
    for (int i = threadIdx.x; i < HW; i += 256) {
        const float v0 = (float)base[((size_t)i << 6) + c0];
        const float v1 = (float)base[((size_t)i << 6) + c0 + 1];
        s  += v0 + v1;
        s2 += v0 * v0 + v1 * v1;
    }
    __shared__ float sh[256], sh2[256];
    sh[threadIdx.x]  = s;
    sh2[threadIdx.x] = s2;
    __syncthreads();
    for (int off = 128; off > 0; off >>= 1) {
        if ((int)threadIdx.x < off) {
            sh[threadIdx.x]  += sh[threadIdx.x + off];
            sh2[threadIdx.x] += sh2[threadIdx.x + off];
        }
        __syncthreads();
    }
    if (threadIdx.x == 0) {
        const float n   = (float)(HW * 2);
        const float m   = sh[0] / n;
        const float var = sh2[0] / n - m * m;
        stats[bg * 2]     = m;
        stats[bg * 2 + 1] = rsqrtf(var + 1e-6f);
    }
}

// ---------------------------------------------------------------------------
// Fused GroupNorm-apply + LeakyReLU(0.2): one thread per (pixel, group)
// ---------------------------------------------------------------------------
__global__ __launch_bounds__(256)
void gn_lrelu_kernel(const _Float16* __restrict__ x, _Float16* __restrict__ y,
                     const float* __restrict__ stats,
                     const float* __restrict__ gamma, const float* __restrict__ beta,
                     int totpix, int HW)
{
    const int i = blockIdx.x * blockDim.x + threadIdx.x;
    if (i >= totpix * 32) return;
    const int g   = i & 31;
    const int pix = i >> 5;
    const int b   = pix / HW;
    const float m = stats[(b * 32 + g) * 2];
    const float r = stats[(b * 32 + g) * 2 + 1];
    const int    c0   = g << 1;
    const size_t base = (((size_t)pix) << 6) + c0;
    float v0 = ((float)x[base]     - m) * r * gamma[c0]     + beta[c0];
    float v1 = ((float)x[base + 1] - m) * r * gamma[c0 + 1] + beta[c0 + 1];
    y[base]     = (_Float16)(v0 >= 0.f ? v0 : 0.2f * v0);
    y[base + 1] = (_Float16)(v1 >= 0.f ? v1 : 0.2f * v1);
}

// ---------------------------------------------------------------------------
__global__ __launch_bounds__(256)
void add_kernel(_Float16* __restrict__ x, const _Float16* __restrict__ h, int n)
{
    const int i = blockIdx.x * blockDim.x + threadIdx.x;
    if (i < n) x[i] = (_Float16)((float)x[i] + (float)h[i]);
}

// ---------------------------------------------------------------------------
// Nearest-neighbor 2x upsample, NHWC; one thread copies 8 channels (16 B)
// ---------------------------------------------------------------------------
__global__ __launch_bounds__(256)
void upsample_kernel(const _Float16* __restrict__ in, _Float16* __restrict__ out, int Hi, int Wi)
{
    const int Ho = Hi * 2, Wo = Wi * 2;
    const size_t tot = (size_t)BATCH * Ho * Wo * 8;
    const size_t i   = (size_t)blockIdx.x * blockDim.x + threadIdx.x;
    if (i >= tot) return;
    const int    ck  = (int)(i & 7);
    const size_t pix = i >> 3;
    const int    wo  = (int)(pix % Wo);
    size_t       t   = pix / Wo;
    const int    ho  = (int)(t % Ho);
    const int    b   = (int)(t / Ho);
    const float4* src = (const float4*)(in + ((((size_t)b * Hi + (ho >> 1)) * Wi + (wo >> 1)) << 6) + (ck << 3));
    *(float4*)(out + ((((size_t)b * Ho + ho) * Wo + wo) << 6) + (ck << 3)) = *src;
}

// ---------------------------------------------------------------------------
// Vector-quantize: nearest of 64 codebook rows; forward value is cb[idx]
// ---------------------------------------------------------------------------
__global__ __launch_bounds__(128)
void vq_kernel(const _Float16* __restrict__ z, const float* __restrict__ cb,
               _Float16* __restrict__ out, int N)
{
    const int n = blockIdx.x * blockDim.x + threadIdx.x;
    if (n >= N) return;
    float zv[CH];
#pragma unroll
    for (int c = 0; c < CH; ++c) zv[c] = (float)z[(size_t)n * CH + c];
    float best = 3.4e38f;
    int   bi   = 0;
    for (int e = 0; e < CH; ++e) {
        float d = 0.f;
#pragma unroll
        for (int c = 0; c < CH; ++c) {
            const float t = zv[c] - cb[e * CH + c];
            d += t * t;
        }
        if (d < best) { best = d; bi = e; }
    }
#pragma unroll
    for (int c = 0; c < CH; ++c) out[(size_t)n * CH + c] = (_Float16)cb[bi * CH + c];
}

// ---------------------------------------------------------------------------
// Layout / dtype conversion kernels
// ---------------------------------------------------------------------------
__global__ __launch_bounds__(256)
void nchw_to_nhwc_kernel(const float* __restrict__ in, _Float16* __restrict__ out, int H, int W)
{
    const size_t tot = (size_t)BATCH * CH * H * W;
    const size_t i   = (size_t)blockIdx.x * blockDim.x + threadIdx.x;
    if (i >= tot) return;
    const int w = (int)(i % W);
    size_t    t = i / W;
    const int h = (int)(t % H);
    t /= H;
    const int c = (int)(t & 63);
    const int b = (int)(t >> 6);
    out[((((size_t)b * H + h) * W + w) << 6) + c] = (_Float16)in[i];
}

__global__ __launch_bounds__(256)
void nhwc_to_nchw_kernel(const _Float16* __restrict__ in, float* __restrict__ out, int H, int W)
{
    const size_t tot = (size_t)BATCH * CH * H * W;
    const size_t i   = (size_t)blockIdx.x * blockDim.x + threadIdx.x;
    if (i >= tot) return;
    const int w = (int)(i % W);
    size_t    t = i / W;
    const int h = (int)(t % H);
    t /= H;
    const int c = (int)(t & 63);
    const int b = (int)(t >> 6);
    out[i] = (float)in[((((size_t)b * H + h) * W + w) << 6) + c];
}

// OIHW f32 -> [O][K] f16 with K = (ky*kw+kx)*64 + ci  (B-operand friendly)
__global__ __launch_bounds__(256)
void wconv_kernel(const float* __restrict__ w, _Float16* __restrict__ wT, int kh, int kw)
{
    const int K = kh * kw * CH;
    const int i = blockIdx.x * blockDim.x + threadIdx.x;
    if (i >= CH * K) return;
    const int o   = i / K;
    const int k   = i - o * K;
    const int ci  = k & 63;
    const int tap = k >> 6;
    const int ky  = tap / kw;
    const int kx  = tap - ky * kw;
    wT[i] = (_Float16)w[(((size_t)o * CH + ci) * kh + ky) * kw + kx];
}

// ---------------------------------------------------------------------------
static inline unsigned cdivz(size_t a, unsigned b) { return (unsigned)((a + b - 1) / b); }

extern "C" void kernel_launch(void* const* d_in, const int* in_sizes, int n_in,
                              void* d_out, int out_size, void* d_ws, size_t ws_size,
                              hipStream_t stream)
{
    (void)in_sizes; (void)n_in; (void)out_size; (void)ws_size;

    const size_t ACT = (size_t)BATCH * 256 * 256 * CH;   // max activation elements
    _Float16* A   = (_Float16*)d_ws;
    _Float16* Bb  = A + ACT;
    _Float16* Cc  = Bb + ACT;
    _Float16* W16 = Cc + ACT;                            // all conv weights, f16 transposed
    float*    stats = (float*)(W16 + (1 << 20));

    // conv table: d_in index of weight, kernel size, offset into W16
    static const int widx[17] = {1, 3, 7, 11, 13, 17, 21, 24, 28, 32, 36, 40, 42, 46, 50, 54, 58};
    static const int wk[17]   = {4, 3, 3, 3, 3, 3, 3, 3, 3, 3, 3, 3, 3, 3, 3, 3, 3};
    size_t woff[17];
    {
        size_t off = 0;
        for (int i = 0; i < 17; ++i) { woff[i] = off; off += (size_t)CH * wk[i] * wk[i] * CH; }
    }
    for (int i = 0; i < 17; ++i) {
        const int tot = CH * wk[i] * wk[i] * CH;
        wconv_kernel<<<cdivz((size_t)tot, 256), 256, 0, stream>>>(
            (const float*)d_in[widx[i]], W16 + woff[i], wk[i], wk[i]);
    }

    auto conv = [&](const _Float16* inp, _Float16* outp, int ci,
                    int Hi, int Wi, int Ho, int Wo, int s, int p) {
        dim3 g(cdivz((size_t)Ho * Wo, TM), BATCH);
        conv_wmma_kernel<<<g, 256, 0, stream>>>(
            inp, W16 + woff[ci], (const float*)d_in[widx[ci] + 1], outp,
            Hi, Wi, Ho, Wo, wk[ci], wk[ci], s, p);
    };
    auto gnstats = [&](const _Float16* x, int HW) {
        gn_stats_kernel<<<BATCH * 32, 256, 0, stream>>>(x, stats, HW);
    };
    auto gnlrelu = [&](const _Float16* x, _Float16* y, int gi, int bi, int HW) {
        const int totpix = BATCH * HW;
        gn_lrelu_kernel<<<cdivz((size_t)totpix * 32, 256), 256, 0, stream>>>(
            x, y, stats, (const float*)d_in[gi], (const float*)d_in[bi], totpix, HW);
    };
    auto addres = [&](_Float16* x, const _Float16* h, int HW) {
        const int tot = BATCH * HW * CH;
        add_kernel<<<cdivz((size_t)tot, 256), 256, 0, stream>>>(x, h, tot);
    };
    auto resblock = [&](_Float16* X, _Float16* T1, _Float16* T2,
                        int g1, int b1, int c1, int g2, int b2, int c2, int H, int W) {
        const int HW = H * W;
        gnstats(X, HW);  gnlrelu(X, T1, g1, b1, HW);
        conv(T1, T2, c1, H, W, H, W, 1, 1);
        gnstats(T2, HW); gnlrelu(T2, T1, g2, b2, HW);
        conv(T1, T2, c2, H, W, H, W, 1, 1);
        addres(X, T2, HW);
    };
    auto upsample = [&](const _Float16* inp, _Float16* outp, int Hi, int Wi) {
        const size_t tot = (size_t)BATCH * (Hi * 2) * (Wi * 2) * 8;
        upsample_kernel<<<cdivz(tot, 256), 256, 0, stream>>>(inp, outp, Hi, Wi);
    };

    // ---- Encoder ----
    nchw_to_nhwc_kernel<<<cdivz(ACT, 256), 256, 0, stream>>>((const float*)d_in[0], A, 256, 256);
    conv(A, Bb, 0, 256, 256, 255, 255, 1, 1);              // in_conv k4 s1 p1
    conv(Bb, A, 1, 255, 255, 128, 128, 2, 1);              // enc0 down
    resblock(A, Bb, Cc, 5, 6, 2, 9, 10, 3, 128, 128);
    conv(A, Bb, 4, 128, 128, 64, 64, 2, 1);                // enc1 down
    resblock(Bb, Cc, A, 15, 16, 5, 19, 20, 6, 64, 64);

    // ---- Vector quantizer (straight-through => forward value is cb[idx]) ----
    {
        const int N = BATCH * 64 * 64;
        vq_kernel<<<cdivz((size_t)N, 128), 128, 0, stream>>>(Bb, (const float*)d_in[23], A, N);
    }

    // ---- Decoder ----
    upsample(A, Bb, 64, 64);
    conv(Bb, A, 7, 128, 128, 128, 128, 1, 1);
    resblock(A, Bb, Cc, 26, 27, 8, 30, 31, 9, 128, 128);
    resblock(A, Bb, Cc, 34, 35, 10, 38, 39, 11, 128, 128);
    upsample(A, Bb, 128, 128);
    conv(Bb, A, 12, 256, 256, 256, 256, 1, 1);
    resblock(A, Bb, Cc, 44, 45, 13, 48, 49, 14, 256, 256);
    resblock(A, Bb, Cc, 52, 53, 15, 56, 57, 16, 256, 256);

    nhwc_to_nchw_kernel<<<cdivz(ACT, 256), 256, 0, stream>>>(A, (float*)d_out, 256, 256);
}
